// ParallelGATBlock_43911745634522
// MI455X (gfx1250) — matrix operand order repaired and checked by reading, not compiled, not run on hardware
//
#include <hip/hip_runtime.h>
#include <hip/hip_bf16.h>
#include <cstdint>

// ---------------------------------------------------------------------------
// ParallelGATBlock for MI455X (gfx1250, wave32)
//   LayerNorm -> fused GAT1/GAT2 feature GEMM (WMMA f16->f32)
//   -> edge softmax (ordered-uint atomicMax + atomicAdd segsum)
//   -> wave-per-edge scatter-add -> ELU -> update GEMM (WMMA, TDM-fed) + residual
// ---------------------------------------------------------------------------

typedef __attribute__((ext_vector_type(16))) _Float16 v16h;
typedef __attribute__((ext_vector_type(8)))  float    v8f;
typedef __attribute__((ext_vector_type(4)))  unsigned int v4u;
typedef __attribute__((ext_vector_type(4)))  int      v4i;
typedef __attribute__((ext_vector_type(8)))  int      v8i;

#define NNODE 40000
#define NCH   128
#define NEDGE 640000
#define NHEAD 12          // 4 heads of 32 + 8 heads of 16
#define NCAT  256

#if defined(__has_builtin)
#if __has_builtin(__builtin_amdgcn_tensor_load_to_lds) && \
    __has_builtin(__builtin_amdgcn_s_wait_tensorcnt)
#define HAVE_TDM 1
#endif
#endif

// ------------------------- small device helpers ----------------------------

__device__ __forceinline__ unsigned f2ord(float f) {
  unsigned u = __float_as_uint(f);
  return (u & 0x80000000u) ? ~u : (u | 0x80000000u);
}
__device__ __forceinline__ float ord2f(unsigned o) {
  return (o & 0x80000000u) ? __uint_as_float(o ^ 0x80000000u)
                           : __uint_as_float(~o);
}
__device__ __forceinline__ float leaky(float v) { return v > 0.f ? v : 0.2f * v; }

// CDNA5 async global->LDS copy (ASYNCcnt), 16 bytes per lane.
__device__ __forceinline__ void async_g2l_b128(uint32_t lds_off, const void* gbase,
                                               uint32_t goff) {
  asm volatile("global_load_async_to_lds_b128 %0, %1, %2"
               :: "v"(lds_off), "v"(goff), "s"(gbase) : "memory");
}
__device__ __forceinline__ void wait_async0() {
  asm volatile("s_wait_asynccnt 0" ::: "memory");
}

// WMMA fragment loaders per CDNA5 ISA 16-bit layouts (CDNA5 ISA).
__device__ __forceinline__ int a_pair_k(int i, int half) {
  return ((i < 4) ? (2 * i) : (16 + 2 * (i - 4))) + 8 * half;
}
template <int W>
__device__ __forceinline__ v16h load_a_frag(const _Float16* shA, int lane, int k0) {
  int m = lane & 15, half = lane >> 4;
  v16h a;
#pragma unroll
  for (int i = 0; i < 8; ++i) {
    int k = k0 + a_pair_k(i, half);
    a[2 * i]     = shA[m * W + k];
    a[2 * i + 1] = shA[m * W + k + 1];
  }
  return a;
}
template <int K>
__device__ __forceinline__ v16h load_b_frag(const _Float16* BT, int col0, int lane,
                                            int k0) {
  int n = lane & 15, half = lane >> 4;
  const _Float16* p = BT + (size_t)(col0 + n) * K;
  v16h b;
#pragma unroll
  for (int i = 0; i < 8; ++i) {
    int k = k0 + a_pair_k(i, half);
    b[2 * i]     = p[k];
    b[2 * i + 1] = p[k + 1];
  }
  return b;
}

// ------------------------------ kernels ------------------------------------

// K1: weights to f16, column-major transposed for contiguous B fragments.
__global__ void k_prep_w(const float* __restrict__ W1, const float* __restrict__ W2,
                         const float* __restrict__ Wu, _Float16* __restrict__ WcT,
                         _Float16* __restrict__ WuT) {
  int id = blockIdx.x * blockDim.x + threadIdx.x;
  if (id < 256 * 128) {
    int j = id / 128, k = id % 128;
    float v = (j < 128) ? W1[k * 128 + j] : W2[k * 128 + (j - 128)];
    WcT[id] = (_Float16)v;
  } else if (id < 2 * 256 * 128) {
    int t = id - 256 * 128;
    int j = t / 256, k = t % 256;
    WuT[t] = (_Float16)Wu[k * 128 + j];
  }
}

// K2: LayerNorm, one wave32 per node (4 channels/lane), output f16.
__global__ void k_layernorm(const float* __restrict__ x, const float* __restrict__ g,
                            const float* __restrict__ b, _Float16* __restrict__ xn16) {
  int wave = threadIdx.x >> 5, lane = threadIdx.x & 31;
  int node = blockIdx.x * 8 + wave;
  if (node >= NNODE) return;
  const float* row = x + (size_t)node * NCH;
  float v[4], s = 0.f;
#pragma unroll
  for (int i = 0; i < 4; ++i) { v[i] = row[lane + 32 * i]; s += v[i]; }
#pragma unroll
  for (int off = 16; off; off >>= 1) s += __shfl_xor(s, off, 32);
  float mu = s * (1.0f / NCH), q = 0.f;
#pragma unroll
  for (int i = 0; i < 4; ++i) { float d = v[i] - mu; q += d * d; }
#pragma unroll
  for (int off = 16; off; off >>= 1) q += __shfl_xor(q, off, 32);
  float rs = rsqrtf(q * (1.0f / NCH) + 1e-5f);
  _Float16* orow = xn16 + (size_t)node * NCH;
#pragma unroll
  for (int i = 0; i < 4; ++i) {
    int c = lane + 32 * i;
    orow[c] = (_Float16)((v[i] - mu) * rs * g[c] + b[c]);
  }
}

// K3: H[N x 256] = xn16[N x 128] @ Wc[128 x 256]   (WMMA f16 -> f32)
// A tile (4 KB) staged with CDNA5 async global->LDS (ASYNCcnt path).
__global__ void k_gemm_h(const _Float16* __restrict__ xn16,
                         const _Float16* __restrict__ WcT, float* __restrict__ H) {
  __shared__ _Float16 shA[16 * 128];
  int t = threadIdx.x;
  int node0 = blockIdx.x * 16;
  {
    uint32_t lds = (uint32_t)(uintptr_t)(&shA[0]) + (uint32_t)t * 16;
    async_g2l_b128(lds, (const void*)(xn16 + (size_t)node0 * NCH), (uint32_t)t * 16);
    wait_async0();
  }
  __syncthreads();
  int wave = t >> 5, lane = t & 31;
#pragma unroll
  for (int sub = 0; sub < 2; ++sub) {
    int col0 = (wave * 2 + sub) * 16;
    v8f c = {};
#pragma unroll
    for (int ks = 0; ks < 4; ++ks) {
      int k0 = ks * 32;
      v16h a = load_a_frag<128>(shA, lane, k0);
      v16h b = load_b_frag<128>(WcT, col0, lane, k0);
      c = __builtin_amdgcn_wmma_f32_16x16x32_f16(false, a, false, b, (short)0, c,
                                                 false, false);
    }
    int n = lane & 15, half = lane >> 4;
#pragma unroll
    for (int j = 0; j < 8; ++j) {
      int row = j + 8 * half;
      H[(size_t)(node0 + row) * NCAT + col0 + n] = c[j];
    }
  }
}

// K4: attention logits per (node, head).
__global__ void k_alpha(const float* __restrict__ H, const float* __restrict__ a1s,
                        const float* __restrict__ a1d, const float* __restrict__ a2s,
                        const float* __restrict__ a2d, float* __restrict__ AS,
                        float* __restrict__ AD) {
  int id = blockIdx.x * blockDim.x + threadIdx.x;
  if (id >= NNODE * NHEAD) return;
  int n = id / NHEAD, hh = id % NHEAD;
  int base, len;
  const float *as, *ad;
  if (hh < 4) { base = hh * 32; len = 32; as = a1s + hh * 32; ad = a1d + hh * 32; }
  else { int h2 = hh - 4; base = 128 + h2 * 16; len = 16; as = a2s + h2 * 16; ad = a2d + h2 * 16; }
  const float* hrow = H + (size_t)n * NCAT + base;
  float s = 0.f, d = 0.f;
  for (int o = 0; o < len; ++o) { float hv = hrow[o]; s += hv * as[o]; d += hv * ad[o]; }
  AS[id] = s;
  AD[id] = d;
}

// K5: init.  ACC=0; segment-max seeded with the self-loop logit; DEN=0.
__global__ void k_init(const float* __restrict__ AS, const float* __restrict__ AD,
                       unsigned* __restrict__ M, float* __restrict__ DEN,
                       float* __restrict__ ACC) {
  int i = blockIdx.x * blockDim.x + threadIdx.x;
  if (i < NNODE * NCAT) ACC[i] = 0.f;
  if (i < NNODE * NHEAD) {
    M[i] = f2ord(leaky(AS[i] + AD[i]));
    DEN[i] = 0.f;
  }
}

// K6: segment max over destinations (ordered-uint atomicMax).
__global__ void k_edge_max(const long long* __restrict__ ei,
                           const float* __restrict__ AS, const float* __restrict__ AD,
                           unsigned* __restrict__ M) {
  int e = blockIdx.x * blockDim.x + threadIdx.x;
  if (e >= NEDGE) return;
  int s = (int)ei[e], d = (int)ei[NEDGE + e];
#pragma unroll
  for (int hh = 0; hh < NHEAD; ++hh) {
    float v = leaky(AS[s * NHEAD + hh] + AD[d * NHEAD + hh]);
    atomicMax(&M[d * NHEAD + hh], f2ord(v));
  }
}

// K7: per-edge exp weights + denominator segsum.
__global__ void k_edge_exp(const long long* __restrict__ ei,
                           const float* __restrict__ AS, const float* __restrict__ AD,
                           const unsigned* __restrict__ M, float* __restrict__ DEN,
                           float* __restrict__ EW) {
  int id = blockIdx.x * blockDim.x + threadIdx.x;
  if (id >= NEDGE * NHEAD) return;
  int e = id / NHEAD, hh = id % NHEAD;
  int s = (int)ei[e], d = (int)ei[NEDGE + e];
  float v = leaky(AS[s * NHEAD + hh] + AD[d * NHEAD + hh]);
  float w = __expf(v - ord2f(M[d * NHEAD + hh]));
  EW[id] = w;
  atomicAdd(&DEN[d * NHEAD + hh], w);
}

// K7b: self-loop weights.
__global__ void k_self_exp(const float* __restrict__ AS, const float* __restrict__ AD,
                           const unsigned* __restrict__ M, float* __restrict__ DEN,
                           float* __restrict__ WSELF) {
  int i = blockIdx.x * blockDim.x + threadIdx.x;
  if (i >= NNODE * NHEAD) return;
  float w = __expf(leaky(AS[i] + AD[i]) - ord2f(M[i]));
  WSELF[i] = w;
  atomicAdd(&DEN[i], w);
}

// K8: message scatter-add.  Wave32 per edge; edge index is wave-uniform
// (scalarizable), each lane gathers 8 channels of H[src] as 2x float4
// (global_load_b128) and issues 8 f32 atomics into ACC[dst] (L2-resident).
__global__ void k_scatter(const long long* __restrict__ ei, const float* __restrict__ H,
                          const float* __restrict__ EW, const float* __restrict__ DEN,
                          float* __restrict__ ACC) {
  int wave = threadIdx.x >> 5, lane = threadIdx.x & 31;
  int e = blockIdx.x * 8 + wave;
  if (e >= NEDGE) return;
  int s = (int)ei[e], d = (int)ei[NEDGE + e];
  int c0 = lane * 8;  // 8 consecutive channels per lane; single head per group
  int hh = (c0 < 128) ? (c0 >> 5) : (4 + ((c0 - 128) >> 4));
  float alpha = EW[(size_t)e * NHEAD + hh] / (DEN[d * NHEAD + hh] + 1e-16f);
  const float4* hp = (const float4*)(H + (size_t)s * NCAT + c0);
  float4 h0 = hp[0], h1 = hp[1];
  float* ap = ACC + (size_t)d * NCAT + c0;
  atomicAdd(ap + 0, h0.x * alpha);
  atomicAdd(ap + 1, h0.y * alpha);
  atomicAdd(ap + 2, h0.z * alpha);
  atomicAdd(ap + 3, h0.w * alpha);
  atomicAdd(ap + 4, h1.x * alpha);
  atomicAdd(ap + 5, h1.y * alpha);
  atomicAdd(ap + 6, h1.z * alpha);
  atomicAdd(ap + 7, h1.w * alpha);
}

// K9: self-loop message + bias + ELU -> f16 concat activations.
__global__ void k_finalize_cat(const float* __restrict__ H, const float* __restrict__ ACC,
                               const float* __restrict__ WSELF,
                               const float* __restrict__ DEN, const float* __restrict__ b1,
                               const float* __restrict__ b2, _Float16* __restrict__ xc16) {
  int id = blockIdx.x * blockDim.x + threadIdx.x;
  if (id >= NNODE * NCAT) return;
  int n = id >> 8, col = id & 255;
  int hh = (col < 128) ? (col >> 5) : (4 + ((col - 128) >> 4));
  float aself = WSELF[n * NHEAD + hh] / (DEN[n * NHEAD + hh] + 1e-16f);
  float v = ACC[id] + H[id] * aself + ((col < 128) ? b1[col] : b2[col - 128]);
  v = v > 0.f ? v : (__expf(v) - 1.f);
  xc16[id] = (_Float16)v;
}

// K10: out = x + xc16 @ Wu + bu   (WMMA, 16-row tile).
// The contiguous 8 KB A tile is fetched by the Tensor Data Mover (TENSORcnt
// path): one D# issued by wave 0, 1024 x 8-byte elements, then
// s_wait_tensorcnt(0) + workgroup barrier.  Falls back to async-LDS copies.
__global__ void k_gemm_out(const _Float16* __restrict__ xc16,
                           const _Float16* __restrict__ WuT, const float* __restrict__ x,
                           const float* __restrict__ bu, float* __restrict__ out) {
  __shared__ _Float16 shA[16 * 256];
  int t = threadIdx.x;
  int node0 = blockIdx.x * 16;
#ifdef HAVE_TDM
  if (t < 32) {  // wave-uniform branch: wave 0 drives the TDM
    uint64_t ga  = (uint64_t)(uintptr_t)(xc16 + (size_t)node0 * NCAT);
    uint32_t lds = (uint32_t)(uintptr_t)(&shA[0]);
    v4u g0;
    g0[0] = 1u;                                            // count=1 (valid D#)
    g0[1] = lds;                                           // lds_addr
    g0[2] = (uint32_t)ga;                                  // global_addr[31:0]
    g0[3] = (uint32_t)((ga >> 32) & 0x1FFFFFFu) | (2u << 30);  // [56:32] | type=2
    v8i g1;
    g1[0] = (3 << 16);           // workgroup_mask=0, data_size=3 (8B elements)
    g1[1] = (int)(1024u << 16);  // tensor_dim0 = 1024 elems  ([79:48] low half)
    g1[2] = (1 << 16);           // tensor_dim1 = 1           ([111:80] low half)
    g1[3] = (int)(1024u << 16);  // tile_dim0   = 1024        ([127:112])
    g1[4] = 1;                   // tile_dim1   = 1           ([143:128])
    g1[5] = 1024;                // tensor_dim0_stride[31:0]
    g1[6] = 0;
    g1[7] = 0;
    v4i z4 = {0, 0, 0, 0};
    v8i z8 = {0, 0, 0, 0, 0, 0, 0, 0};
    // 6-arg therock-10.0 signature: (g0, g1, g2, g3, g4, cpol)
    __builtin_amdgcn_tensor_load_to_lds(g0, g1, z4, z4, z8, 0);
    __builtin_amdgcn_s_wait_tensorcnt(0);
  }
#else
  {
    uint32_t lbase = (uint32_t)(uintptr_t)(&shA[0]);
    const void* gbase = (const void*)(xc16 + (size_t)node0 * NCAT);
    async_g2l_b128(lbase + (uint32_t)t * 32, gbase, (uint32_t)t * 32);
    async_g2l_b128(lbase + (uint32_t)t * 32 + 16, gbase, (uint32_t)t * 32 + 16);
    wait_async0();
  }
#endif
  __syncthreads();
  int wave = t >> 5, lane = t & 31;
  int col0 = wave * 16;
  v8f c = {};
#pragma unroll
  for (int ks = 0; ks < 8; ++ks) {
    int k0 = ks * 32;
    v16h a = load_a_frag<256>(shA, lane, k0);
    v16h b = load_b_frag<256>(WuT, col0, lane, k0);
    c = __builtin_amdgcn_wmma_f32_16x16x32_f16(false, a, false, b, (short)0, c, false,
                                               false);
  }
  int n = lane & 15, half = lane >> 4;
#pragma unroll
  for (int j = 0; j < 8; ++j) {
    int row = j + 8 * half;
    size_t idx = (size_t)(node0 + row) * NCH + col0 + n;
    out[idx] = c[j] + x[idx] + bu[col0 + n];
  }
}

// ------------------------------- launcher ----------------------------------

extern "C" void kernel_launch(void* const* d_in, const int* in_sizes, int n_in,
                              void* d_out, int out_size, void* d_ws, size_t ws_size,
                              hipStream_t stream) {
  const float*     x    = (const float*)d_in[0];
  const long long* ei   = (const long long*)d_in[1];
  const float*     ln_g = (const float*)d_in[2];
  const float*     ln_b = (const float*)d_in[3];
  const float*     W1   = (const float*)d_in[4];
  const float*     a1s  = (const float*)d_in[5];
  const float*     a1d  = (const float*)d_in[6];
  const float*     b1   = (const float*)d_in[7];
  const float*     W2   = (const float*)d_in[8];
  const float*     a2s  = (const float*)d_in[9];
  const float*     a2d  = (const float*)d_in[10];
  const float*     b2   = (const float*)d_in[11];
  const float*     Wu   = (const float*)d_in[12];
  const float*     bu   = (const float*)d_in[13];
  float*           out  = (float*)d_out;

  // Workspace carve (~153 MB total).
  size_t off = 0;
  char* base = (char*)d_ws;
  auto carve = [&](size_t bytes) -> void* {
    void* p = base + off;
    off += (bytes + 255) & ~(size_t)255;
    return p;
  };
  _Float16* xn16  = (_Float16*)carve((size_t)NNODE * NCH * 2);
  _Float16* WcT   = (_Float16*)carve((size_t)256 * 128 * 2);
  _Float16* WuT   = (_Float16*)carve((size_t)128 * 256 * 2);
  float*    H     = (float*)carve((size_t)NNODE * NCAT * 4);
  float*    AS    = (float*)carve((size_t)NNODE * NHEAD * 4);
  float*    AD    = (float*)carve((size_t)NNODE * NHEAD * 4);
  unsigned* M     = (unsigned*)carve((size_t)NNODE * NHEAD * 4);
  float*    DEN   = (float*)carve((size_t)NNODE * NHEAD * 4);
  float*    WSELF = (float*)carve((size_t)NNODE * NHEAD * 4);
  float*    EW    = (float*)carve((size_t)NEDGE * NHEAD * 4);
  float*    ACC   = (float*)carve((size_t)NNODE * NCAT * 4);
  _Float16* xc16  = (_Float16*)carve((size_t)NNODE * NCAT * 2);

  const int T = 256;
  k_prep_w<<<(2 * 256 * 128 + T - 1) / T, T, 0, stream>>>(W1, W2, Wu, WcT, WuT);
  k_layernorm<<<NNODE / 8, T, 0, stream>>>(x, ln_g, ln_b, xn16);
  k_gemm_h<<<NNODE / 16, T, 0, stream>>>(xn16, WcT, H);
  k_alpha<<<(NNODE * NHEAD + T - 1) / T, T, 0, stream>>>(H, a1s, a1d, a2s, a2d, AS, AD);
  k_init<<<(NNODE * NCAT + T - 1) / T, T, 0, stream>>>(AS, AD, M, DEN, ACC);
  k_edge_max<<<(NEDGE + T - 1) / T, T, 0, stream>>>(ei, AS, AD, M);
  k_edge_exp<<<(NEDGE * NHEAD + T - 1) / T, T, 0, stream>>>(ei, AS, AD, M, DEN, EW);
  k_self_exp<<<(NNODE * NHEAD + T - 1) / T, T, 0, stream>>>(AS, AD, M, DEN, WSELF);
  k_scatter<<<(NEDGE + 7) / 8, T, 0, stream>>>(ei, H, EW, DEN, ACC);
  k_finalize_cat<<<(NNODE * NCAT + T - 1) / T, T, 0, stream>>>(H, ACC, WSELF, DEN, b1,
                                                               b2, xc16);
  k_gemm_out<<<NNODE / 16, T, 0, stream>>>(xc16, WuT, x, bu, out);
}
